// GNNML3_64991445123387
// MI455X (gfx1250) — matrix operand order repaired and checked by reading, not compiled, not run on hardware
//
#include <hip/hip_runtime.h>

#define NNODES 50000
#define NEDGES 1600000
#define NGRAPH 128
#define NSUP 5
#define NIN 16
#define HID 64
#define KOUT (NSUP * HID)   // 320
#define NCLS 6
#define BN_EPS 1e-5f

typedef __attribute__((ext_vector_type(16))) _Float16 v16h;
typedef __attribute__((ext_vector_type(8)))  _Float16 v8h;
typedef __attribute__((ext_vector_type(8)))  float    v8f;

// ---------------------------------------------------------------------------
// Weight prep: W [NSUP, fin, HID] f32  ->  Wt [KOUT, 64] f16, K-major,
// zero-padded to K=64 (so layer 1 with fin=16 uses the same GEMM).
// Column n = ksup*64 + h ; Wt[n*64 + k] = W[ksup][k][h]
// ---------------------------------------------------------------------------
__global__ void prep_weights_k(const float* __restrict__ W, int fin,
                               _Float16* __restrict__ Wt) {
  int idx = blockIdx.x * blockDim.x + threadIdx.x;
  if (idx >= KOUT * HID) return;
  int n = idx >> 6;       // 0..319
  int k = idx & 63;       // 0..63
  int ksup = n >> 6;
  int h = n & 63;
  float v = (k < fin) ? W[(ksup * fin + k) * HID + h] : 0.0f;
  Wt[n * HID + k] = (_Float16)v;
}

// Hh [N,64] f16: first NIN cols = x, rest zero
__global__ void init_h_k(const float* __restrict__ x, _Float16* __restrict__ Hh) {
  int idx = blockIdx.x * blockDim.x + threadIdx.x;
  if (idx >= NNODES * HID) return;
  int n = idx >> 6, c = idx & 63;
  Hh[idx] = (c < NIN) ? (_Float16)x[n * NIN + c] : (_Float16)0.0f;
}

// ---------------------------------------------------------------------------
// WMMA GEMM: Y[N, 320] (f16) = Hh[N, 64] (f16) @ Wt (K-major f16), f32 accum.
// One wave per 16x16 tile; K loop = 2 x v_wmma_f32_16x16x32_f16.
// A frag per ISA 16-bit A(16x32): lanes<16 K 0..7 / 16..23, lanes>=16 +8.
// B frag same K order as A: lanes<16 K 0..7 / 16..23, lanes>=16 +8.
// ---------------------------------------------------------------------------
__global__ __launch_bounds__(256) void gemm_wmma_k(
    const _Float16* __restrict__ Hh, const _Float16* __restrict__ Wt,
    _Float16* __restrict__ Y) {
  const int TN = KOUT / 16;                       // 20 column tiles
  const int total = (NNODES / 16) * TN;           // 62500 tiles
  int wave = blockIdx.x * 8 + (threadIdx.x >> 5); // wave-uniform
  if (wave >= total) return;                      // EXEC all-ones inside
  int lane = threadIdx.x & 31;
  int m0 = (wave / TN) * 16;
  int n0 = (wave % TN) * 16;
  int hi  = lane >> 4;
  int l15 = lane & 15;

  const _Float16* arow = Hh + (m0 + l15) * HID + (hi ? 8 : 0);
  const _Float16* bcol = Wt + (n0 + l15) * HID + (hi ? 8 : 0);

  v8f c = {0.f, 0.f, 0.f, 0.f, 0.f, 0.f, 0.f, 0.f};
#pragma unroll
  for (int kb = 0; kb < HID; kb += 32) {
    v8h alo = *(const v8h*)(arow + kb);
    v8h ahi = *(const v8h*)(arow + kb + 16);
    v8h blo = *(const v8h*)(bcol + kb);
    v8h bhi = *(const v8h*)(bcol + kb + 16);
    v16h a = __builtin_shufflevector(alo, ahi, 0,1,2,3,4,5,6,7,8,9,10,11,12,13,14,15);
    v16h b = __builtin_shufflevector(blo, bhi, 0,1,2,3,4,5,6,7,8,9,10,11,12,13,14,15);
    c = __builtin_amdgcn_wmma_f32_16x16x32_f16(false, a, false, b,
                                               (short)0, c, false, false);
  }

  int col = n0 + l15;
  int rb  = m0 + (hi ? 8 : 0);
#pragma unroll
  for (int r = 0; r < 8; r++)
    Y[(rb + r) * KOUT + col] = (_Float16)c[r];
}

// acc[n*64+c] = bias[c]
__global__ void init_acc_k(const float* __restrict__ b, float* __restrict__ acc) {
  int idx = blockIdx.x * blockDim.x + threadIdx.x;
  if (idx >= NNODES * HID) return;
  acc[idx] = b[idx & 63];
}

// Edge scatter: acc[dst,c] += sum_k ea[e,k] * Y[src, k*64+c].
// 64 consecutive threads share one edge -> coalesced Y loads & atomics;
// Y (32 MB) and acc (12.8 MB) are L2-resident on MI455X (192 MB L2).
__global__ void edge_scatter_k(const int* __restrict__ src,
                               const int* __restrict__ dst,
                               const float* __restrict__ ea,
                               const _Float16* __restrict__ Y,
                               float* __restrict__ acc) {
  int idx = blockIdx.x * blockDim.x + threadIdx.x;
  if (idx >= NEDGES * HID) return;
  int e = idx >> 6, c = idx & 63;
  int s = src[e], d = dst[e];
  const _Float16* yp = Y + s * KOUT + c;
  const float* ep = ea + e * NSUP;
  float v = 0.0f;
#pragma unroll
  for (int k = 0; k < NSUP; k++)
    v += ep[k] * (float)yp[k * HID];
  atomicAdd(acc + d * HID + c, v);
}

// Hh = relu(acc) as f16 (input to next GEMM / pooling)
__global__ void relu_h_k(const float* __restrict__ acc, _Float16* __restrict__ Hh) {
  int idx = blockIdx.x * blockDim.x + threadIdx.x;
  if (idx >= NNODES * HID) return;
  float v = acc[idx];
  Hh[idx] = (_Float16)(v > 0.0f ? v : 0.0f);
}

__global__ void pool_init_k(float* __restrict__ pool) {
  int idx = blockIdx.x * blockDim.x + threadIdx.x;
  if (idx < NGRAPH * 2 * HID) pool[idx] = 0.0f;  // bits(0)==0.0f for max too
}

// pooled[g, 0:64] = segment_sum, pooled[g, 64:128] = segment_max (values >= 0)
__global__ void pool_scatter_k(const _Float16* __restrict__ Hh,
                               const int* __restrict__ batch,
                               float* __restrict__ pool) {
  int idx = blockIdx.x * blockDim.x + threadIdx.x;
  if (idx >= NNODES * HID) return;
  int n = idx >> 6, c = idx & 63;
  int g = batch[n];
  float v = (float)Hh[idx];
  atomicAdd(pool + g * (2 * HID) + c, v);
  // post-ReLU v >= 0: uint bit pattern is order-preserving
  atomicMax((unsigned int*)(pool + g * (2 * HID) + HID + c), __float_as_uint(v));
}

// BN + FC + log_softmax. One wave32 per graph; shfl_xor reductions.
__global__ __launch_bounds__(32) void head_k(
    const float* __restrict__ pool, const float* __restrict__ gamma,
    const float* __restrict__ beta, const float* __restrict__ mean,
    const float* __restrict__ var, const float* __restrict__ fcw,
    const float* __restrict__ fcb, float* __restrict__ out) {
  int g = blockIdx.x;
  int lane = threadIdx.x;
  float l[NCLS];
#pragma unroll
  for (int cc = 0; cc < NCLS; cc++) l[cc] = 0.0f;
#pragma unroll
  for (int j = 0; j < 4; j++) {
    int ch = lane + 32 * j;
    float p = pool[g * (2 * HID) + ch];
    float bnv = (p - mean[ch]) * rsqrtf(var[ch] + BN_EPS) * gamma[ch] + beta[ch];
#pragma unroll
    for (int cc = 0; cc < NCLS; cc++) l[cc] += bnv * fcw[ch * NCLS + cc];
  }
#pragma unroll
  for (int off = 16; off >= 1; off >>= 1)
#pragma unroll
    for (int cc = 0; cc < NCLS; cc++) l[cc] += __shfl_xor(l[cc], off, 32);
  if (lane == 0) {
    float logit[NCLS], m = -3.4e38f;
#pragma unroll
    for (int cc = 0; cc < NCLS; cc++) {
      logit[cc] = l[cc] + fcb[cc];
      m = fmaxf(m, logit[cc]);
    }
    float se = 0.0f;
#pragma unroll
    for (int cc = 0; cc < NCLS; cc++) se += expf(logit[cc] - m);
    float lse = m + logf(se);
#pragma unroll
    for (int cc = 0; cc < NCLS; cc++) out[g * NCLS + cc] = logit[cc] - lse;
  }
}

static inline int nb(long n) { return (int)((n + 255) / 256); }

extern "C" void kernel_launch(void* const* d_in, const int* in_sizes, int n_in,
                              void* d_out, int out_size, void* d_ws, size_t ws_size,
                              hipStream_t stream) {
  const float* x      = (const float*)d_in[0];
  const int*   eidx   = (const int*)d_in[1];
  const float* eattr  = (const float*)d_in[2];
  const int*   batch  = (const int*)d_in[3];
  const float* W[4]   = {(const float*)d_in[4], (const float*)d_in[6],
                         (const float*)d_in[8], (const float*)d_in[10]};
  const float* bias[4]= {(const float*)d_in[5], (const float*)d_in[7],
                         (const float*)d_in[9], (const float*)d_in[11]};
  const float* gamma  = (const float*)d_in[12];
  const float* beta   = (const float*)d_in[13];
  const float* mean   = (const float*)d_in[14];
  const float* var    = (const float*)d_in[15];
  const float* fcw    = (const float*)d_in[16];
  const float* fcb    = (const float*)d_in[17];
  float* out = (float*)d_out;

  const int* src = eidx;           // edge_index[0]
  const int* dst = eidx + NEDGES;  // edge_index[1]

  // Workspace layout (all offsets 256B-aligned)
  char* ws = (char*)d_ws;
  _Float16* Hh  = (_Float16*)(ws);                         //  6,400,000 B
  _Float16* Y   = (_Float16*)(ws + 6400000);               // 32,000,000 B
  float*    acc = (float*)   (ws + 38400000);              // 12,800,000 B
  _Float16* Wt0 = (_Float16*)(ws + 51200000);              // 4 x 40,960 B
  float*    pool= (float*)   (ws + 51200000 + 4 * 40960);  //     65,536 B

  // Weight prep (padded K=64, K-major, f16)
  for (int layer = 0; layer < 4; layer++) {
    int fin = (layer == 0) ? NIN : HID;
    prep_weights_k<<<nb(KOUT * HID), 256, 0, stream>>>(
        W[layer], fin, Wt0 + (size_t)layer * KOUT * HID);
  }
  init_h_k<<<nb((long)NNODES * HID), 256, 0, stream>>>(x, Hh);

  const int gemm_waves = (NNODES / 16) * (KOUT / 16);  // 62500
  for (int layer = 0; layer < 4; layer++) {
    gemm_wmma_k<<<(gemm_waves + 7) / 8, 256, 0, stream>>>(
        Hh, Wt0 + (size_t)layer * KOUT * HID, Y);
    init_acc_k<<<nb((long)NNODES * HID), 256, 0, stream>>>(bias[layer], acc);
    edge_scatter_k<<<nb((long)NEDGES * HID), 256, 0, stream>>>(
        src, dst, eattr, Y, acc);
    relu_h_k<<<nb((long)NNODES * HID), 256, 0, stream>>>(acc, Hh);
  }

  pool_init_k<<<nb(NGRAPH * 2 * HID), 256, 0, stream>>>(pool);
  pool_scatter_k<<<nb((long)NNODES * HID), 256, 0, stream>>>(Hh, batch, pool);
  head_k<<<NGRAPH, 32, 0, stream>>>(pool, gamma, beta, mean, var, fcw, fcb, out);
}
